// Qwen3_5MoeGatedDeltaNet_44547400794831
// MI455X (gfx1250) — matrix operand order repaired and checked by reading, not compiled, not run on hardware
//
#include <hip/hip_runtime.h>
#include <hip/hip_bf16.h>
#include <math.h>

// Problem constants (reference: B=2,S=2048,D=2048, HK=16,HV=32,DK=DV=128)
#define B_     2
#define S_     2048
#define D_     2048
#define HK_    16
#define HV_    32
#define DK_    128
#define DV_    128
#define KDIM_  2048
#define VDIM_  4096
#define CONVD_ 8192           // 2*KDIM + VDIM
#define LDMIX_ 12288          // fused GEMM width: CONV_DIM(8192) + VDIM(4096, z)
#define KSIZE_ 4

typedef __attribute__((ext_vector_type(16))) __bf16 v16bf;
typedef __attribute__((ext_vector_type(8)))  float  v8f;

__device__ __forceinline__ float sigmoidf_(float x) { return 1.f / (1.f + __expf(-x)); }

// ---------------- fp32 -> bf16 convert ----------------
__global__ void cvt_f32_bf16(const float* __restrict__ in, __bf16* __restrict__ out, long n) {
  long i = (long)blockIdx.x * blockDim.x + threadIdx.x;
  if (i < n) out[i] = (__bf16)in[i];
}

// ---------------- fp32 (K x N) -> bf16 transposed (N x K) ----------------
__global__ void transpose_f32_bf16(const float* __restrict__ w, __bf16* __restrict__ wt,
                                   int K, int N) {
  long i = (long)blockIdx.x * blockDim.x + threadIdx.x;
  if (i < (long)K * N) {
    int k = (int)(i / N);
    int n = (int)(i % N);
    wt[(long)n * K + k] = (__bf16)w[i];
  }
}

// ---------------- bf16 WMMA GEMM: C(MxN,f32) = A(MxK) * Bt(NxK)^T ----------------
// Block tile 128x128, K-step 32, 8 waves arranged 4(M) x 2(N), each wave 2x4 WMMA tiles.
__global__ __launch_bounds__(256) void gemm_bf16_wmma(
    const __bf16* __restrict__ A, const __bf16* __restrict__ Bt,
    float* __restrict__ C, int M, int N, int K) {
  (void)M;
  __shared__ __align__(16) __bf16 As[128 * 40];  // 32 cols + 8 pad
  __shared__ __align__(16) __bf16 Bs[128 * 40];

  const int tid  = threadIdx.x;
  const int lane = tid & 31;
  const int wv   = tid >> 5;
  const int wm   = wv & 3;        // 0..3  -> 32-row strip
  const int wn   = wv >> 2;       // 0..1  -> 64-col strip
  const int idx  = lane & 15;     // M (A) / N (B,C) index
  const int hl   = lane >> 4;     // half selector

  const int m0 = blockIdx.y * 128;
  const int n0 = blockIdx.x * 128;

  v8f acc[2][4];
  #pragma unroll
  for (int tm = 0; tm < 2; ++tm)
    #pragma unroll
    for (int tn = 0; tn < 4; ++tn)
      #pragma unroll
      for (int j = 0; j < 8; ++j) acc[tm][tn][j] = 0.f;

  const int ldr = tid >> 1;             // 0..127 row within tile
  const int ldc = (tid & 1) * 16;       // 0 or 16
  const __bf16* Ag = A  + (size_t)(m0 + ldr) * K + ldc;
  const __bf16* Bg = Bt + (size_t)(n0 + ldr) * K + ldc;

  for (int k0 = 0; k0 < K; k0 += 32) {
    uint4 a0 = *(const uint4*)(Ag);
    uint4 a1 = *(const uint4*)(Ag + 8);
    uint4 b0 = *(const uint4*)(Bg);
    uint4 b1 = *(const uint4*)(Bg + 8);
    if (k0 + 64 <= K) {                 // prefetch next K-tile (global_prefetch_b8)
      __builtin_prefetch(Ag + 32, 0, 1);
      __builtin_prefetch(Bg + 32, 0, 1);
    }
    __syncthreads();
    *(uint4*)(As + ldr * 40 + ldc)     = a0;
    *(uint4*)(As + ldr * 40 + ldc + 8) = a1;
    *(uint4*)(Bs + ldr * 40 + ldc)     = b0;
    *(uint4*)(Bs + ldr * 40 + ldc + 8) = b1;
    __syncthreads();

    // A fragment: lane (idx,hl): elems 0..7 -> k = 8*hl..8*hl+7 ; 8..15 -> k = 16+8*hl..
    v16bf afr[2], bfr[4];
    #pragma unroll
    for (int tm = 0; tm < 2; ++tm) {
      const __bf16* p = As + (wm * 32 + tm * 16 + idx) * 40 + hl * 8;
      ((uint4*)&afr[tm])[0] = *(const uint4*)(p);
      ((uint4*)&afr[tm])[1] = *(const uint4*)(p + 16);
    }
    // B fragment: lane (idx,hl): elems 0..7 -> k = 8*hl..8*hl+7 ; 8..15 -> k = 16+8*hl..
    #pragma unroll
    for (int tn = 0; tn < 4; ++tn) {
      const __bf16* p = Bs + (wn * 64 + tn * 16 + idx) * 40 + hl * 8;
      ((uint4*)&bfr[tn])[0] = *(const uint4*)(p);
      ((uint4*)&bfr[tn])[1] = *(const uint4*)(p + 16);
    }
    #pragma unroll
    for (int tm = 0; tm < 2; ++tm)
      #pragma unroll
      for (int tn = 0; tn < 4; ++tn)
        acc[tm][tn] = __builtin_amdgcn_wmma_f32_16x16x32_bf16(
            false, afr[tm], false, bfr[tn], (short)0, acc[tm][tn], false, false);

    Ag += 32;
    Bg += 32;
  }

  // C/D layout: lane (idx,hl): n = idx, m = j + 8*hl
  #pragma unroll
  for (int tm = 0; tm < 2; ++tm)
    #pragma unroll
    for (int tn = 0; tn < 4; ++tn) {
      int mb = m0 + wm * 32 + tm * 16 + hl * 8;
      int nb = n0 + wn * 64 + tn * 16 + idx;
      float* cp = C + (size_t)mb * N + nb;
      #pragma unroll
      for (int j = 0; j < 8; ++j) cp[(size_t)j * N] = acc[tm][tn][j];
    }
}

// ---------------- b/a projections + beta, g (one wave per token; HV==32==warpSize) ----
__global__ __launch_bounds__(256) void proj_ba(
    const float* __restrict__ hid, const float* __restrict__ wb, const float* __restrict__ wa,
    const float* __restrict__ dtb, const float* __restrict__ alog,
    float* __restrict__ gout, float* __restrict__ betaout) {
  int lane = threadIdx.x & 31;
  int wv   = threadIdx.x >> 5;
  long bs  = (long)blockIdx.x * 8 + wv;
  const float* h = hid + bs * D_;
  float accb = 0.f, acca = 0.f;
  for (int k = 0; k < D_; ++k) {
    float hv = h[k];
    accb = fmaf(hv, wb[k * HV_ + lane], accb);
    acca = fmaf(hv, wa[k * HV_ + lane], acca);
  }
  betaout[bs * HV_ + lane] = sigmoidf_(accb);
  float x  = acca + dtb[lane];
  float sp = (x > 20.f) ? x : log1pf(__expf(x));
  gout[bs * HV_ + lane] = -__expf(alog[lane]) * sp;
}

// ---------------- causal conv1d(K=4) + SiLU + per-head L2 norm (q,k) ----------------
__global__ __launch_bounds__(128) void conv_silu_norm(
    const float* __restrict__ mixed, const float* __restrict__ convw,
    float* __restrict__ qb, float* __restrict__ kb, float* __restrict__ vb) {
  __shared__ float red[128];
  int t = threadIdx.x, slice = blockIdx.x, s = blockIdx.y, b = blockIdx.z;
  int c = slice * 128 + t;
  long rowb = (long)b * S_;
  float acc = 0.f;
  #pragma unroll
  for (int j = 0; j < KSIZE_; ++j) {
    int sj = s - (KSIZE_ - 1) + j;
    if (sj >= 0) acc = fmaf(mixed[(rowb + sj) * (long)LDMIX_ + c], convw[c * KSIZE_ + j], acc);
  }
  float y  = acc * sigmoidf_(acc);
  long bs  = rowb + s;
  if (slice < 32) {                     // q or k head-slice: L2 normalize over 128
    red[t] = y * y;
    __syncthreads();
    for (int off = 64; off > 0; off >>= 1) {
      if (t < off) red[t] += red[t + off];
      __syncthreads();
    }
    float out = y * rsqrtf(red[0] + 1e-6f);
    if (slice < 16) qb[bs * KDIM_ + slice * DK_ + t] = out;
    else            kb[bs * KDIM_ + (slice - 16) * DK_ + t] = out;
  } else {
    vb[bs * VDIM_ + (slice - 32) * DV_ + t] = y;
  }
}

// ---------------- gated delta rule scan: state 128x128 f32 in registers ----------------
// One workgroup per (b,h). Thread t: column c=t&127, rows [64*(t>>7), +64).
__global__ __launch_bounds__(256) void gated_delta_scan(
    const float* __restrict__ qb, const float* __restrict__ kb, const float* __restrict__ vb,
    const float* __restrict__ gb, const float* __restrict__ betab, float* __restrict__ core) {
  __shared__ float lq[128], lk[128], lv[128], ldl[128], lred[256], lsc[2];
  int t = threadIdx.x;
  int c = t & 127;
  int hf = t >> 7;
  int h = blockIdx.x, b = blockIdx.y;
  int hq = h >> 1;                                        // jnp.repeat(..., 2, axis=2)
  const float* qp = qb + ((long)b * S_ * HK_ + hq) * DK_;
  const float* kp = kb + ((long)b * S_ * HK_ + hq) * DK_;
  const float* vp = vb + ((long)b * S_ * HV_ + h) * DV_;
  const float* gp = gb + (long)b * S_ * HV_ + h;
  const float* bp = betab + (long)b * S_ * HV_ + h;
  float* op = core + ((long)b * S_ * HV_ + h) * DV_;

  float Sst[64];
  #pragma unroll
  for (int i = 0; i < 64; ++i) Sst[i] = 0.f;

  for (int s = 0; s < S_; ++s) {
    if (t < 128) { lq[t] = qp[t] * 0.08838834764831845f; lk[t] = kp[t]; }
    else         { lv[t - 128] = vp[t - 128]; }
    if (t == 0)  { lsc[0] = gp[0]; lsc[1] = bp[0]; }
    __syncthreads();
    float eg  = __expf(lsc[0]);
    float bet = lsc[1];
    const float* kr = lk + hf * 64;
    const float* qr = lq + hf * 64;
    float part = 0.f;
    #pragma unroll
    for (int i = 0; i < 64; ++i) {                 // decay + v_old = k . S
      float sv = Sst[i] * eg;
      Sst[i] = sv;
      part = fmaf(kr[i], sv, part);
    }
    lred[t] = part;
    __syncthreads();
    if (t < 128) ldl[t] = (lv[t] - (lred[t] + lred[t + 128])) * bet;  // delta
    __syncthreads();
    float d = ldl[c];
    float part2 = 0.f;
    #pragma unroll
    for (int i = 0; i < 64; ++i) {                 // S += k (x) delta ; o = q . S
      float sv = fmaf(kr[i], d, Sst[i]);
      Sst[i] = sv;
      part2 = fmaf(qr[i], sv, part2);
    }
    lred[t] = part2;
    __syncthreads();
    if (t < 128) op[t] = lred[t] + lred[t + 128];
    __syncthreads();
    qp += HK_ * DK_; kp += HK_ * DK_; vp += HV_ * DV_; gp += HV_; bp += HV_; op += HV_ * DV_;
  }
}

// ---------------- RMS norm over DV * norm_weight * SiLU(z) -> bf16 ----------------
__global__ __launch_bounds__(128) void norm_gate(
    const float* __restrict__ core, const float* __restrict__ mixed,
    const float* __restrict__ nw, __bf16* __restrict__ corebf) {
  __shared__ float red[128];
  int t = threadIdx.x, h = blockIdx.x, s = blockIdx.y, b = blockIdx.z;
  long bs = (long)b * S_ + s;
  float cv = core[bs * VDIM_ + h * DV_ + t];
  red[t] = cv * cv;
  __syncthreads();
  for (int off = 64; off > 0; off >>= 1) {
    if (t < off) red[t] += red[t + off];
    __syncthreads();
  }
  float sc = rsqrtf(red[0] * (1.f / DV_) + 1e-6f);
  float zv = mixed[bs * (long)LDMIX_ + CONVD_ + h * DV_ + t];
  float o  = cv * sc * nw[t] * (zv * sigmoidf_(zv));
  corebf[bs * VDIM_ + h * DV_ + t] = (__bf16)o;
}

extern "C" void kernel_launch(void* const* d_in, const int* in_sizes, int n_in,
                              void* d_out, int out_size, void* d_ws, size_t ws_size,
                              hipStream_t stream) {
  (void)in_sizes; (void)n_in; (void)out_size; (void)ws_size;
  const float* hidden = (const float*)d_in[0];
  const float* w_qkv  = (const float*)d_in[1];
  const float* w_z    = (const float*)d_in[2];
  const float* w_b    = (const float*)d_in[3];
  const float* w_a    = (const float*)d_in[4];
  const float* w_out  = (const float*)d_in[5];
  const float* conv_w = (const float*)d_in[6];
  const float* dt_b   = (const float*)d_in[7];
  const float* A_log  = (const float*)d_in[8];
  const float* normw  = (const float*)d_in[9];
  float* out = (float*)d_out;

  char* ws = (char*)d_ws;
  size_t off = 0;
  auto alloc = [&](size_t bytes) {
    void* p = ws + off;
    off += (bytes + 255) & ~(size_t)255;
    return p;
  };
  const long MS = (long)B_ * S_;                                  // 4096 tokens
  __bf16* hid_bf  = (__bf16*)alloc((size_t)MS * D_ * 2);
  __bf16* wcatT   = (__bf16*)alloc((size_t)LDMIX_ * D_ * 2);       // [w_qkv|w_z]^T
  __bf16* woutT   = (__bf16*)alloc((size_t)D_ * VDIM_ * 2);
  float*  mixed   = (float*)alloc((size_t)MS * LDMIX_ * 4);
  float*  qbuf    = (float*)alloc((size_t)MS * KDIM_ * 4);
  float*  kbuf    = (float*)alloc((size_t)MS * KDIM_ * 4);
  float*  vbuf    = (float*)alloc((size_t)MS * VDIM_ * 4);
  float*  gbuf    = (float*)alloc((size_t)MS * HV_ * 4);
  float*  betabuf = (float*)alloc((size_t)MS * HV_ * 4);
  float*  corebuf = (float*)alloc((size_t)MS * VDIM_ * 4);
  __bf16* corebf  = (__bf16*)alloc((size_t)MS * VDIM_ * 2);

  { long n = MS * D_;
    cvt_f32_bf16<<<(unsigned)((n + 255) / 256), 256, 0, stream>>>(hidden, hid_bf, n); }
  { long n = (long)D_ * CONVD_;
    transpose_f32_bf16<<<(unsigned)((n + 255) / 256), 256, 0, stream>>>(w_qkv, wcatT, D_, CONVD_); }
  { long n = (long)D_ * VDIM_;
    transpose_f32_bf16<<<(unsigned)((n + 255) / 256), 256, 0, stream>>>(
        w_z, wcatT + (size_t)CONVD_ * D_, D_, VDIM_); }
  { long n = (long)VDIM_ * D_;
    transpose_f32_bf16<<<(unsigned)((n + 255) / 256), 256, 0, stream>>>(w_out, woutT, VDIM_, D_); }

  // Fused projection GEMM: mixed(4096 x 12288) = hidden * [w_qkv | w_z]
  gemm_bf16_wmma<<<dim3(LDMIX_ / 128, (unsigned)(MS / 128)), 256, 0, stream>>>(
      hid_bf, wcatT, mixed, (int)MS, LDMIX_, D_);

  proj_ba<<<(unsigned)(MS / 8), 256, 0, stream>>>(hidden, w_b, w_a, dt_b, A_log, gbuf, betabuf);

  conv_silu_norm<<<dim3(64, S_, B_), 128, 0, stream>>>(mixed, conv_w, qbuf, kbuf, vbuf);

  gated_delta_scan<<<dim3(HV_, B_), 256, 0, stream>>>(qbuf, kbuf, vbuf, gbuf, betabuf, corebuf);

  norm_gate<<<dim3(HV_, S_, B_), 128, 0, stream>>>(corebuf, mixed, normw, corebf);

  // Output GEMM: out(4096 x 2048) = core * w_out
  gemm_bf16_wmma<<<dim3(D_ / 128, (unsigned)(MS / 128)), 256, 0, stream>>>(
      corebf, woutT, out, (int)MS, D_, VDIM_);
}
